// LightningDecoderLayer_25640954757695
// MI455X (gfx1250) — matrix operand order repaired and checked by reading, not compiled, not run on hardware
//
#include <hip/hip_runtime.h>
#include <hip/hip_bf16.h>
#include <stdint.h>

#define H_   16
#define DH_  64
#define T_   1024
#define B_   2
#define HID_ 1024
#define M_   (B_ * T_)   // 2048 token rows

typedef __attribute__((ext_vector_type(16))) __bf16 v16bf;
typedef __attribute__((ext_vector_type(8)))  __bf16 v8bf;
typedef __attribute__((ext_vector_type(8)))  float  v8f;

// ---------------------------------------------------------------- conversions
__global__ void cvt_bf16_kernel(const float* __restrict__ in, __bf16* __restrict__ out, int n) {
    int i = blockIdx.x * 256 + threadIdx.x;
    if (i < n) out[i] = (__bf16)in[i];
}

// in: K x N row-major (fp32), out: N x K row-major (bf16) == transposed
__global__ void cvt_bf16_t_kernel(const float* __restrict__ in, __bf16* __restrict__ out,
                                  int K, int N) {
    int i = blockIdx.x * 256 + threadIdx.x;
    if (i < K * N) {
        int k = i / N, n = i % N;
        out[(size_t)n * K + k] = (__bf16)in[i];
    }
}

// ---------------------------------------------------------------- WMMA GEMM
// C[M,N] (fp32) = A[M,K] (bf16 row-major) @ B (via Bt[N,K] bf16 row-major)
// block = 256 threads = 8 waves (2 waveM x 4 waveN), wave tile 16x64,
// block tile 32x256. A tile staged to LDS via async global->LDS, double
// buffered with ASYNCcnt pipelining; B fragments loaded direct (L2 resident).
#define AROW_ 40   // padded LDS row stride in bf16 (80B = 20-bank stride)
__global__ void gemm_bf16_wmma_kernel(const __bf16* __restrict__ A,
                                      const __bf16* __restrict__ Bt,
                                      float* __restrict__ C,
                                      int M, int N, int Kd) {
    const int lane   = threadIdx.x & 31;
    const int wave   = threadIdx.x >> 5;
    const int waveM  = wave >> 2;
    const int waveN  = wave & 3;
    const int blockM = blockIdx.y * 32;
    const int tileM  = blockM + waveM * 16;
    const int tileN  = blockIdx.x * 256 + waveN * 64;
    const int mn     = lane & 15;
    const int half   = lane >> 4;

    __shared__ __bf16 ldsA[2][32 * AROW_];

    // async loader mapping: waves 0..3 stage A[blockM..+31][kb..kb+31]
    const int f    = wave * 32 + lane;     // 0..255, loaders f<128
    const int lrow = f >> 2;
    const int lq   = f & 3;
    const __bf16* gsrc = A + (size_t)(blockM + lrow) * Kd + lq * 8;
    const uint32_t ldsdst0 = (uint32_t)(uintptr_t)(&ldsA[0][lrow * AROW_ + lq * 8]);
    const uint32_t ldsdst1 = (uint32_t)(uintptr_t)(&ldsA[1][lrow * AROW_ + lq * 8]);

    const __bf16* __restrict__ brow0 = Bt + (size_t)(tileN +  0 + mn) * Kd;
    const __bf16* __restrict__ brow1 = Bt + (size_t)(tileN + 16 + mn) * Kd;
    const __bf16* __restrict__ brow2 = Bt + (size_t)(tileN + 32 + mn) * Kd;
    const __bf16* __restrict__ brow3 = Bt + (size_t)(tileN + 48 + mn) * Kd;

    v8f c0 = {}, c1 = {}, c2 = {}, c3 = {};

    if (f < 128) {
        asm volatile("global_load_async_to_lds_b128 %0, %1, off"
                     :: "v"(ldsdst0),
                        "v"((unsigned long long)(uintptr_t)gsrc)
                     : "memory");
    }

    int buf = 0;
    for (int kb = 0; kb < Kd; kb += 32) {
        const int next = kb + 32;
        if (f < 128) {
            if (next < Kd) {
                asm volatile("global_load_async_to_lds_b128 %0, %1, off"
                             :: "v"(buf ? ldsdst0 : ldsdst1),
                                "v"((unsigned long long)(uintptr_t)(gsrc + next))
                             : "memory");
                asm volatile("s_wait_asynccnt 0x1" ::: "memory");
            } else {
                asm volatile("s_wait_asynccnt 0x0" ::: "memory");
            }
        }
        __syncthreads();   // staged A tile visible to all 8 waves

        // A 16x32 fragment from LDS: values 0..7 -> K=8*half+j, 8..15 -> +16
        const __bf16* aL = &ldsA[buf][(waveM * 16 + mn) * AROW_];
        v8bf a0 = *(const v8bf*)(aL + half * 8);
        v8bf a1 = *(const v8bf*)(aL + 16 + half * 8);
        v16bf a;
#pragma unroll
        for (int j = 0; j < 8; ++j) { a[j] = a0[j]; a[8 + j] = a1[j]; }

        // 4 WMMAs sharing the A fragment (wave tile 16x64)
#define GEMM_STEP(BROW, ACC)                                                   \
        {                                                                      \
            v8bf b0 = *(const v8bf*)(BROW + kb + half * 8);                    \
            v8bf b1 = *(const v8bf*)(BROW + kb + 16 + half * 8);               \
            v16bf b;                                                           \
            _Pragma("unroll")                                                  \
            for (int j = 0; j < 8; ++j) { b[j] = b0[j]; b[8 + j] = b1[j]; }    \
            ACC = __builtin_amdgcn_wmma_f32_16x16x32_bf16(                     \
                      false, a, false, b, (short)0, ACC, false, false);        \
        }
        GEMM_STEP(brow0, c0)
        GEMM_STEP(brow1, c1)
        GEMM_STEP(brow2, c2)
        GEMM_STEP(brow3, c3)
#undef GEMM_STEP

        __syncthreads();   // everyone done reading buf before it is refilled
        buf ^= 1;
    }

    // C/D layout: VGPR r -> row r (lanes 0-15) / r+8 (lanes 16-31), col = mn
#pragma unroll
    for (int r = 0; r < 8; ++r) {
        const size_t row = (size_t)(tileM + r + 8 * half) * N;
        C[row + tileN +  0 + mn] = c0[r];
        C[row + tileN + 16 + mn] = c1[r];
        C[row + tileN + 32 + mn] = c2[r];
        C[row + tileN + 48 + mn] = c3[r];
    }
}

// ------------------------------------------- causal dwconv + SiLU + RoPE + L2
// grid = B*T*H blocks, 64 threads (one per head dim d)
__global__ void conv_rope_norm_kernel(const float* __restrict__ pre,
                                      const float* __restrict__ cw,
                                      const float* __restrict__ cb,
                                      float* __restrict__ out,
                                      int doRope, int doNorm) {
    const int idx = blockIdx.x;
    const int h   = idx & (H_ - 1);
    const int bt  = idx >> 4;
    const int t   = bt & (T_ - 1);
    const int d   = threadIdx.x;
    const int cch = h * DH_ + d;

    float y = cb[cch];
#pragma unroll
    for (int j = 0; j < 4; ++j) {
        int tt = t - 3 + j;
        if (tt >= 0) y += pre[(size_t)(bt - 3 + j) * (H_ * DH_) + cch] * cw[cch * 4 + j];
    }
    float s = y / (1.f + expf(-y));   // SiLU

    __shared__ float sh[DH_];
    __shared__ float red[DH_];
    sh[d] = s;
    __syncthreads();

    float val;
    if (doRope) {
        int j  = (d < 32) ? d : d - 32;
        int fi = (2 * j) & 31;
        float ang = (float)t * __powf(10000.f, -(2.f * (float)fi) / 64.f);
        float cs = cosf(ang), sn = sinf(ang);
        float x1 = sh[2 * j], x2 = sh[2 * j + 1];
        val = (d < 32) ? (x1 * cs - x2 * sn) : (x1 * sn + x2 * cs);
    } else {
        val = s;
    }

    if (doNorm) {
        red[d] = val * val;
        __syncthreads();
        for (int off = 32; off; off >>= 1) {
            if (d < off) red[d] += red[d + off];
            __syncthreads();
        }
        val *= rsqrtf(red[0] + 1e-6f);
    }
    out[(size_t)idx * DH_ + d] = val;   // layout [b,t,h,d]
}

// ---------------------------------------------------------------- alpha/beta
__global__ void alphabeta_kernel(const float* __restrict__ x,
                                 const float* __restrict__ bw, const float* __restrict__ bb,
                                 const float* __restrict__ gkw, const float* __restrict__ gkb,
                                 const float* __restrict__ A_log, const float* __restrict__ dt_bias,
                                 float* __restrict__ alpha, float* __restrict__ beta) {
    const int bt = blockIdx.x;
    const float* xr = x + (size_t)bt * HID_;
    const int h = threadIdx.x >> 4;
    const int s = threadIdx.x & 15;
    float pb = 0.f, pg = 0.f;
    for (int i = s; i < HID_; i += 16) {
        float xv = xr[i];
        pb += xv * bw[i * H_ + h];
        pg += xv * gkw[i * H_ + h];
    }
    __shared__ float rb[H_][16], rg[H_][16];
    rb[h][s] = pb; rg[h][s] = pg;
    __syncthreads();
    for (int off = 8; off; off >>= 1) {
        if (s < off) { rb[h][s] += rb[h][s + off]; rg[h][s] += rg[h][s + off]; }
        __syncthreads();
    }
    if (s == 0) {
        float db = rb[h][0] + bb[h];
        beta[(size_t)bt * H_ + h] = 1.f / (1.f + expf(-db));
        float dg = rg[h][0] + gkb[h] + dt_bias[h];
        float sp = (dg > 20.f) ? dg : log1pf(expf(dg));
        alpha[(size_t)bt * H_ + h] = expf(-expf(A_log[h]) * sp);
    }
}

// ---------------------------------------------------------------- delta rule
// grid = B*H blocks, 256 threads: thread (g,e), g = 16-row slice of state
// column e. S[d][e] fp32 in LDS (stride 65, conflict free).
__global__ void delta_rule_kernel(const float* __restrict__ q, const float* __restrict__ k,
                                  const float* __restrict__ v,
                                  const float* __restrict__ alpha, const float* __restrict__ beta,
                                  const float* __restrict__ D_param,
                                  float* __restrict__ o) {
    const int b = blockIdx.x >> 4;
    const int h = blockIdx.x & (H_ - 1);
    const int e = threadIdx.x & 63;
    const int g = threadIdx.x >> 6;       // 0..3
    const int d0 = g * 16;

    __shared__ float S[DH_ * 65];
    __shared__ float qs[DH_], ks[DH_], vs[DH_];
    __shared__ float pq[4][65], pk[4][65];
    __shared__ float pqk[4];

#pragma unroll
    for (int dd = 0; dd < 16; ++dd) S[(d0 + dd) * 65 + e] = 0.f;
    const float Dp = D_param[h];
    __syncthreads();

    for (int t = 0; t < T_; ++t) {
        const size_t base = (((size_t)b * T_ + t) * H_ + h) * DH_;
        if (g == 0)      qs[e] = q[base + e];
        else if (g == 1) ks[e] = k[base + e];
        else if (g == 2) vs[e] = v[base + e];
        __syncthreads();

        float qS = 0.f, kS = 0.f, qk = 0.f;
#pragma unroll
        for (int dd = 0; dd < 16; ++dd) {
            const int d = d0 + dd;
            const float Sde = S[d * 65 + e];
            qS += qs[d] * Sde;
            kS += ks[d] * Sde;
            qk += qs[d] * ks[d];
        }
        pq[g][e] = qS;
        pk[g][e] = kS;
        if (e == 0) pqk[g] = qk;
        __syncthreads();

        const float al = alpha[((size_t)b * T_ + t) * H_ + h];
        const float be = beta[((size_t)b * T_ + t) * H_ + h];
        const float kT = pk[0][e] + pk[1][e] + pk[2][e] + pk[3][e];
        if (g == 0) {
            const float qT  = pq[0][e] + pq[1][e] + pq[2][e] + pq[3][e];
            const float qkT = pqk[0] + pqk[1] + pqk[2] + pqk[3];
            o[base + e] = qT + Dp * qkT * vs[e];
        }
        const float coef = be * vs[e] - al * be * kT;
#pragma unroll
        for (int dd = 0; dd < 16; ++dd) {
            const int d = d0 + dd;
            S[d * 65 + e] = al * S[d * 65 + e] + ks[d] * coef;
        }
        __syncthreads();
    }
}

// ------------------------------------------------- RMS-norm + gate -> bf16
__global__ void gate_out_kernel(const float* __restrict__ o, const float* __restrict__ pre_g,
                                const float* __restrict__ onorm_w, __bf16* __restrict__ act) {
    const int idx = blockIdx.x;
    const int h   = idx & (H_ - 1);
    const int bt  = idx >> 4;
    const int d   = threadIdx.x;
    const float ov = o[(size_t)idx * DH_ + d];
    __shared__ float red[DH_];
    red[d] = ov * ov;
    __syncthreads();
    for (int off = 32; off; off >>= 1) {
        if (d < off) red[d] += red[d + off];
        __syncthreads();
    }
    float on  = ov * rsqrtf(red[0] * (1.f / DH_) + 1e-6f) * onorm_w[d];
    float g   = pre_g[(size_t)bt * (H_ * DH_) + h * DH_ + d];
    float val = g * (on / (1.f + expf(-on)));
    act[(size_t)bt * (H_ * DH_) + h * DH_ + d] = (__bf16)val;
}

// ---------------------------------------------------------------- launch
extern "C" void kernel_launch(void* const* d_in, const int* in_sizes, int n_in,
                              void* d_out, int out_size, void* d_ws, size_t ws_size,
                              hipStream_t stream) {
    const float* x       = (const float*)d_in[0];
    const float* Wq      = (const float*)d_in[1];
    const float* Wk      = (const float*)d_in[2];
    const float* Wv      = (const float*)d_in[3];
    const float* Wg      = (const float*)d_in[4];
    const float* Wo      = (const float*)d_in[5];
    const float* bw      = (const float*)d_in[6];
    const float* bb      = (const float*)d_in[7];
    const float* gkw     = (const float*)d_in[8];
    const float* gkb     = (const float*)d_in[9];
    const float* qcw     = (const float*)d_in[10];
    const float* qcb     = (const float*)d_in[11];
    const float* kcw     = (const float*)d_in[12];
    const float* kcb     = (const float*)d_in[13];
    const float* vcw     = (const float*)d_in[14];
    const float* vcb     = (const float*)d_in[15];
    const float* A_log   = (const float*)d_in[16];
    const float* D_param = (const float*)d_in[17];
    const float* dt_bias = (const float*)d_in[18];
    const float* onorm_w = (const float*)d_in[19];
    float* out = (float*)d_out;

    char* base = (char*)d_ws;
    auto alloc = [&](size_t bytes) {
        char* p = base;
        base += (bytes + 255) & ~(size_t)255;
        return p;
    };
    const size_t NTOK = (size_t)M_ * HID_;
    __bf16* xb    = (__bf16*)alloc(NTOK * 2);
    __bf16* Wqt   = (__bf16*)alloc((size_t)HID_ * HID_ * 2);
    __bf16* Wkt   = (__bf16*)alloc((size_t)HID_ * HID_ * 2);
    __bf16* Wvt   = (__bf16*)alloc((size_t)HID_ * HID_ * 2);
    __bf16* Wgt   = (__bf16*)alloc((size_t)HID_ * HID_ * 2);
    __bf16* Wot   = (__bf16*)alloc((size_t)HID_ * HID_ * 2);
    float*  pre_q = (float*)alloc(NTOK * 4);
    float*  pre_k = (float*)alloc(NTOK * 4);
    float*  pre_v = (float*)alloc(NTOK * 4);
    float*  pre_g = (float*)alloc(NTOK * 4);
    float*  qbuf  = (float*)alloc(NTOK * 4);
    float*  kbuf  = (float*)alloc(NTOK * 4);
    float*  vbuf  = (float*)alloc(NTOK * 4);
    float*  alph  = (float*)alloc((size_t)M_ * H_ * 4);
    float*  betb  = (float*)alloc((size_t)M_ * H_ * 4);
    float*  obuf  = pre_q;              // dead after conv-q
    __bf16* act   = (__bf16*)pre_k;     // dead after conv-k

    const int nTok = (int)NTOK;
    const int nW   = HID_ * HID_;

    // 1) precision staging for tensor cores
    cvt_bf16_kernel<<<(nTok + 255) / 256, 256, 0, stream>>>(x, xb, nTok);
    cvt_bf16_t_kernel<<<(nW + 255) / 256, 256, 0, stream>>>(Wq, Wqt, HID_, HID_);
    cvt_bf16_t_kernel<<<(nW + 255) / 256, 256, 0, stream>>>(Wk, Wkt, HID_, HID_);
    cvt_bf16_t_kernel<<<(nW + 255) / 256, 256, 0, stream>>>(Wv, Wvt, HID_, HID_);
    cvt_bf16_t_kernel<<<(nW + 255) / 256, 256, 0, stream>>>(Wg, Wgt, HID_, HID_);
    cvt_bf16_t_kernel<<<(nW + 255) / 256, 256, 0, stream>>>(Wo, Wot, HID_, HID_);

    // 2) WMMA projection GEMMs (2048x1024x1024 each), async-LDS staged A
    dim3 ggrid(HID_ / 256, M_ / 32);
    gemm_bf16_wmma_kernel<<<ggrid, 256, 0, stream>>>(xb, Wqt, pre_q, M_, HID_, HID_);
    gemm_bf16_wmma_kernel<<<ggrid, 256, 0, stream>>>(xb, Wkt, pre_k, M_, HID_, HID_);
    gemm_bf16_wmma_kernel<<<ggrid, 256, 0, stream>>>(xb, Wvt, pre_v, M_, HID_, HID_);
    gemm_bf16_wmma_kernel<<<ggrid, 256, 0, stream>>>(xb, Wgt, pre_g, M_, HID_, HID_);

    // 3) dwconv + SiLU (+RoPE +L2 for q/k)
    const int nBTH = B_ * T_ * H_;
    conv_rope_norm_kernel<<<nBTH, DH_, 0, stream>>>(pre_q, qcw, qcb, qbuf, 1, 1);
    conv_rope_norm_kernel<<<nBTH, DH_, 0, stream>>>(pre_k, kcw, kcb, kbuf, 1, 1);
    conv_rope_norm_kernel<<<nBTH, DH_, 0, stream>>>(pre_v, vcw, vcb, vbuf, 0, 0);

    // 4) gating scalars
    alphabeta_kernel<<<M_, 256, 0, stream>>>(x, bw, bb, gkw, gkb, A_log, dt_bias, alph, betb);

    // 5) sequential delta-rule scan (fp32, LDS-resident 64x64 state)
    delta_rule_kernel<<<B_ * H_, 256, 0, stream>>>(qbuf, kbuf, vbuf, alph, betb, D_param, obuf);

    // 6) RMS-norm + gate, emit bf16 activations
    gate_out_kernel<<<nBTH, DH_, 0, stream>>>(obuf, pre_g, onorm_w, act);

    // 7) output projection straight into d_out (fp32)
    gemm_bf16_wmma_kernel<<<ggrid, 256, 0, stream>>>(act, Wot, out, M_, HID_, HID_);
}
